// conv1_max_embedding_45354854645903
// MI455X (gfx1250) — matrix operand order repaired and checked by reading, not compiled, not run on hardware
//
#include <hip/hip_runtime.h>

#define NPTS  500000
#define CIN   256
#define CCONV 512
#define BATCH 64
#define D1    1024
#define D2    256
#define BNEPS 1e-5f
#define NCHUNK 7813            // ceil(500000 / 64) rows per chunk

typedef __bf16 bf16x16 __attribute__((ext_vector_type(16)));
typedef float  f32x8   __attribute__((ext_vector_type(8)));
typedef float  f32x2   __attribute__((ext_vector_type(2)));
typedef unsigned u32x4v __attribute__((ext_vector_type(4)));
typedef int      i32x4v __attribute__((ext_vector_type(4)));
typedef int      i32x8v __attribute__((ext_vector_type(8)));

// ---- helpers ---------------------------------------------------------------
__device__ __forceinline__ unsigned short f2bf(float f) {     // RNE f32 -> bf16 bits
  unsigned u = __float_as_uint(f);
  u += 0x7fffu + ((u >> 16) & 1u);
  return (unsigned short)(u >> 16);
}
// order-preserving float -> uint key (so max reduces via native u32 atomics)
__device__ __forceinline__ unsigned fkey(float f) {
  unsigned b = __float_as_uint(f);
  return (b & 0x80000000u) ? ~b : (b | 0x80000000u);
}
__device__ __forceinline__ float fdec(unsigned k) {
  unsigned b = (k & 0x80000000u) ? (k & 0x7fffffffu) : ~k;
  return __uint_as_float(b);
}
#define KEY_NEG_INF 0x007FFFFFu   // fkey(-inf)

// ---- B-stage path selection ------------------------------------------------
#if __has_builtin(__builtin_amdgcn_tensor_load_to_lds)
#define BPATH_TDM 1
#elif __has_builtin(__builtin_amdgcn_global_load_async_to_lds_b128)
#define BPATH_ASYNC 1
#else
#define BPATH_PLAIN 1
#endif

#if defined(BPATH_TDM)
// One TDM descriptor: 1-D tile of 8192 x 4B elements (32 KB) global -> LDS.
__device__ __forceinline__ void tdm_copy_32k(const unsigned* gsrc, unsigned* ldst) {
  unsigned long long ga = (unsigned long long)gsrc;
  u32x4v g0 = { 1u,                                        // count=1, user mode
                (unsigned)(unsigned long long)ldst,        // lds byte address
                (unsigned)ga,                              // global_addr[31:0]
                ((unsigned)(ga >> 32) & 0x01FFFFFFu) | 0x80000000u }; // [56:32] | type=2
  i32x8v g1 = { 0x00020000,            // data_size = 4 B, no mask/flags
                (int)(8192u << 16),    // tensor_dim0 = 8192 (bits 79:48)
                (int)(1u << 16),       // tensor_dim1 = 1    (bits 111:80)
                (int)(8192u << 16),    // tile_dim0  = 8192  (bits 127:112)
                1,                     // tile_dim1 = 1, tile_dim2 = 0
                8192,                  // tensor_dim0_stride
                0, 0 };
  i32x4v z4 = { 0, 0, 0, 0 };
#if __clang_major__ >= 23
  i32x8v z8 = { 0, 0, 0, 0, 0, 0, 0, 0 };
  __builtin_amdgcn_tensor_load_to_lds(g0, g1, z4, z4, z8, 0);
#else
  __builtin_amdgcn_tensor_load_to_lds(g0, g1, z4, z4, 0);
#endif
}
#endif

#if defined(BPATH_ASYNC)
__device__ __forceinline__ void async_copy_8k(const unsigned* gsrc, unsigned* ldst, int tid) {
  const __attribute__((address_space(1))) unsigned* g =
      (const __attribute__((address_space(1))) unsigned*)gsrc;
  __attribute__((address_space(3))) unsigned* l =
      (__attribute__((address_space(3))) unsigned*)ldst;
  for (int i = tid * 4; i < 8192; i += 1024)
    __builtin_amdgcn_global_load_async_to_lds_b128(g + i, l + i, 0, 0);
}
#endif

__device__ __forceinline__ void stage_b(const unsigned* __restrict__ wHi,
                                        const unsigned* __restrict__ wLo,
                                        unsigned* bh, unsigned* bl,
                                        int ks, int tid, int wid) {
#if defined(BPATH_TDM)
  (void)tid;
  if (wid == 0)      tdm_copy_32k(wHi + ks * 8192, bh);
  else if (wid == 1) tdm_copy_32k(wLo + ks * 8192, bl);
#elif defined(BPATH_ASYNC)
  (void)wid;
  async_copy_8k(wHi + ks * 8192, bh, tid);
  async_copy_8k(wLo + ks * 8192, bl, tid);
#else
  (void)wid;
  for (int i = tid; i < 8192; i += 256) {
    bh[i] = wHi[ks * 8192 + i];
    bl[i] = wLo[ks * 8192 + i];
  }
#endif
}

// ensure buffer for step `ks` is complete (next step may already be in flight)
__device__ __forceinline__ void wait_b(int ks, int wid) {
#if defined(BPATH_TDM)
  if (wid < 2) {
    if (ks < 7) __builtin_amdgcn_s_wait_tensorcnt(1);
    else        __builtin_amdgcn_s_wait_tensorcnt(0);
  }
#elif defined(BPATH_ASYNC)
  (void)wid;
#if __has_builtin(__builtin_amdgcn_s_wait_asynccnt)
  if (ks < 7) __builtin_amdgcn_s_wait_asynccnt(16);
  else        __builtin_amdgcn_s_wait_asynccnt(0);
#else
  if (ks < 7) asm volatile("s_wait_asynccnt 16" ::: "memory");
  else        asm volatile("s_wait_asynccnt 0" ::: "memory");
#endif
#else
  (void)ks; (void)wid;      // plain path: dscnt flushed by __syncthreads()
#endif
}

// ---- K0: convert W_conv to bf16 hi/lo in WMMA B-fragment order; init keys ---
// fragment entry idx = ((ks*32+t)*32 + lane)*8 + v ; 65536 entries total
__global__ void k0_prep(const float* __restrict__ Wc, unsigned* __restrict__ wHi,
                        unsigned* __restrict__ wLo, unsigned* __restrict__ gKey) {
  int idx  = blockIdx.x * 256 + threadIdx.x;
  int v    = idx & 7;
  int lane = (idx >> 3) & 31;
  int f    = idx >> 8;               // 0..255 : ks*32 + t
  int ks   = f >> 5, t = f & 31;
  int K    = ks * 32 + ((lane & 16) ? 16 : 0) + 2 * v;   // B: lanes0-15 K=0..15
  int Nc   = t * 16 + (lane & 15);
  float w0 = Wc[K * CCONV + Nc];
  float w1 = Wc[(K + 1) * CCONV + Nc];
  unsigned short h0 = f2bf(w0), h1 = f2bf(w1);
  unsigned short l0 = f2bf(w0 - __uint_as_float((unsigned)h0 << 16));
  unsigned short l1 = f2bf(w1 - __uint_as_float((unsigned)h1 << 16));
  wHi[idx] = (unsigned)h0 | ((unsigned)h1 << 16);
  wLo[idx] = (unsigned)l0 | ((unsigned)l1 << 16);
  if (idx < BATCH * CCONV) gKey[idx] = KEY_NEG_INF;      // re-init every launch
}

// ---- K1: fused  x = feats@W_conv + b  --> segment-max into pooled keys -----
// persistent blocks; chunk = 64 rows x 512 channels; split-bf16 (3 WMMAs/tile);
// B double-buffered in LDS, staged by TDM one k-step ahead.
__global__ void __launch_bounds__(256) gemm_pool(
    const float* __restrict__ feats, const int* __restrict__ bidx,
    const float* __restrict__ bconv,
    const unsigned* __restrict__ wHi, const unsigned* __restrict__ wLo,
    unsigned* __restrict__ gKey) {
  extern __shared__ __align__(32) unsigned smem[];
  unsigned* pk    = smem;               // 64*512 pooled keys        (128 KB)
  unsigned* bBase = smem + 32768;       // 2 x (bH 8192 | bL 8192)   (128 KB)
  unsigned* aH    = smem + 65536;       // A hi fragments (64 rows)  (  4 KB)
  unsigned* aL    = aH + 1024;          // A lo fragments            (  4 KB)

  const int tid  = threadIdx.x;
  const int lane = tid & 31, wid = tid >> 5;
  const int rt   = wid & 3;            // row-tile (0..3) within 64-row chunk
  const int colh = wid >> 2;           // channel half (0..1), 16 n-tiles each

  for (int i = tid; i < BATCH * CCONV; i += 256) pk[i] = KEY_NEG_INF;
  __syncthreads();

  for (int c = blockIdx.x; c < NCHUNK; c += gridDim.x) {
    f32x8 acc[16] = {};
    stage_b(wHi, wLo, bBase, bBase + 8192, 0, tid, wid);  // prologue: ks=0 -> buf0
    for (int ks = 0; ks < 8; ++ks) {                      // K = 8 steps of 32
      __syncthreads();   // prev compute done: A and buf[(ks+1)&1] reusable
      if (ks < 7) {
        unsigned* nb = bBase + ((ks + 1) & 1) * 16384;
        stage_b(wHi, wLo, nb, nb + 8192, ks + 1, tid, wid);
      }
      // A fill: load f32, split to hi/lo bf16, store in A-fragment layout
      for (int task = tid; task < 1024; task += 256) {
        int m = task >> 4, p = task & 15;                 // row-in-chunk, K-pair
        int row = c * 64 + m; if (row > NPTS - 1) row = NPTS - 1;
        const float* fp = feats + (long)row * CIN + ks * 32 + 2 * p;
        float f0 = fp[0], f1 = fp[1];
        unsigned short h0 = f2bf(f0), h1 = f2bf(f1);
        unsigned short l0 = f2bf(f0 - __uint_as_float((unsigned)h0 << 16));
        unsigned short l1 = f2bf(f1 - __uint_as_float((unsigned)h1 << 16));
        int mm = m & 15, rtt = m >> 4;
        int ln = mm + ((p >> 3) ? 16 : 0);                // A: lanes16-31 K=16..31
        int v  = p & 7;
        int o  = (rtt * 32 + ln) * 8 + v;
        aH[o] = (unsigned)h0 | ((unsigned)h1 << 16);
        aL[o] = (unsigned)l0 | ((unsigned)l1 << 16);
        if (ks == 0) {                                    // prefetch next chunk
          int nr = (c + (int)gridDim.x) * 64 + m; if (nr > NPTS - 1) nr = NPTS - 1;
          __builtin_prefetch(feats + (long)nr * CIN + 2 * p, 0, 1);
        }
      }
      wait_b(ks, wid);                                    // buf[ks&1] arrived
      __syncthreads();
      const unsigned* bH = bBase + (ks & 1) * 16384;
      const unsigned* bL = bH + 8192;
      bf16x16 avh = *(const bf16x16*)(aH + (rt * 32 + lane) * 8);
      bf16x16 avl = *(const bf16x16*)(aL + (rt * 32 + lane) * 8);
#pragma unroll
      for (int j = 0; j < 16; ++j) {
        int t = colh * 16 + j;
        bf16x16 bvh = *(const bf16x16*)(bH + (t * 32 + lane) * 8);
        bf16x16 bvl = *(const bf16x16*)(bL + (t * 32 + lane) * 8);
        acc[j] = __builtin_amdgcn_wmma_f32_16x16x32_bf16(false, avh, false, bvh, (short)0, acc[j], false, false);
        acc[j] = __builtin_amdgcn_wmma_f32_16x16x32_bf16(false, avh, false, bvl, (short)0, acc[j], false, false);
        acc[j] = __builtin_amdgcn_wmma_f32_16x16x32_bf16(false, avl, false, bvh, (short)0, acc[j], false, false);
      }
    }
    // epilogue: +bias, segment-max into LDS pooled keys (ds_max_u32)
    int base = c * 64 + rt * 16 + ((lane & 16) ? 8 : 0);  // D: lanes16-31 -> M+8
    int bi[8];
#pragma unroll
    for (int i = 0; i < 8; ++i) {
      int r = base + i; if (r > NPTS - 1) r = NPTS - 1;   // dup of last row: max-idempotent
      bi[i] = bidx[r];
    }
#pragma unroll
    for (int j = 0; j < 16; ++j) {
      int t  = colh * 16 + j;
      int ch = t * 16 + (lane & 15);
      float bias = bconv[ch];
      const float* av = (const float*)&acc[j];
#pragma unroll
      for (int i = 0; i < 8; ++i)
        atomicMax(&pk[bi[i] * CCONV + ch], fkey(av[i] + bias));
    }
  }
  __syncthreads();
  for (int i = tid; i < BATCH * CCONV; i += 256)          // one flush per block
    atomicMax(&gKey[i], pk[i]);
}

// ---- K2: h = pooled @ W1 + b1  (exact fp32 WMMA, 16x16x4) ------------------
__global__ void fc1_wmma(const unsigned* __restrict__ gKey, const float* __restrict__ W1,
                         const float* __restrict__ b1, float* __restrict__ h) {
  int gw = blockIdx.x * 8 + (threadIdx.x >> 5);   // 256 waves: rt 0..3, nt 0..63
  int lane = threadIdx.x & 31;
  int rt = gw & 3, nt = gw >> 2;
  int row  = rt * 16 + (lane & 15);
  int Ncol = nt * 16 + (lane & 15);
  f32x8 acc = {};
  for (int k = 0; k < CCONV; k += 4) {
    int kb = k + ((lane & 16) ? 2 : 0);
    f32x2 a, b;
    a.x = fdec(gKey[row * CCONV + kb]);           // decode pooled keys inline
    a.y = fdec(gKey[row * CCONV + kb + 1]);
    b.x = W1[kb * D1 + Ncol];
    b.y = W1[(kb + 1) * D1 + Ncol];
    acc = __builtin_amdgcn_wmma_f32_16x16x4_f32(false, a, false, b, (short)0, acc, false, false);
  }
  const float* av = (const float*)&acc;
  float bb = b1[Ncol];
#pragma unroll
  for (int i = 0; i < 8; ++i) {
    int r = rt * 16 + i + ((lane & 16) ? 8 : 0);
    h[r * D1 + Ncol] = av[i] + bb;
  }
}

// ---- K3: training-mode BN (biased var) + ReLU ------------------------------
__global__ void bn_relu(const float* __restrict__ h, const float* __restrict__ gamma,
                        const float* __restrict__ beta, float* __restrict__ h2) {
  int col = blockIdx.x * 256 + threadIdx.x;       // 1024 columns
  float s = 0.f, sq = 0.f;
  for (int r = 0; r < BATCH; ++r) { float v = h[r * D1 + col]; s += v; sq += v * v; }
  float mean = s * (1.f / BATCH);
  float var  = sq * (1.f / BATCH) - mean * mean;
  float g = gamma[col] * rsqrtf(var + BNEPS), bt = beta[col];
  for (int r = 0; r < BATCH; ++r) {
    float v = (h[r * D1 + col] - mean) * g + bt;
    h2[r * D1 + col] = v > 0.f ? v : 0.f;
  }
}

// ---- K4: out = h2 @ W2 + b2 ------------------------------------------------
__global__ void fc2_wmma(const float* __restrict__ h2, const float* __restrict__ W2,
                         const float* __restrict__ b2, float* __restrict__ out) {
  int gw = blockIdx.x * 8 + (threadIdx.x >> 5);   // 64 waves: rt 0..3, nt 0..15
  int lane = threadIdx.x & 31;
  int rt = gw & 3, nt = gw >> 2;
  int row  = rt * 16 + (lane & 15);
  int Ncol = nt * 16 + (lane & 15);
  f32x8 acc = {};
  for (int k = 0; k < D1; k += 4) {
    int kb = k + ((lane & 16) ? 2 : 0);
    f32x2 a, b;
    a.x = h2[row * D1 + kb];
    a.y = h2[row * D1 + kb + 1];
    b.x = W2[kb * D2 + Ncol];
    b.y = W2[(kb + 1) * D2 + Ncol];
    acc = __builtin_amdgcn_wmma_f32_16x16x4_f32(false, a, false, b, (short)0, acc, false, false);
  }
  const float* av = (const float*)&acc;
  float bb = b2[Ncol];
#pragma unroll
  for (int i = 0; i < 8; ++i) {
    int r = rt * 16 + i + ((lane & 16) ? 8 : 0);
    out[r * D2 + Ncol] = av[i] + bb;
  }
}

// ---- host-side orchestration ----------------------------------------------
extern "C" void kernel_launch(void* const* d_in, const int* in_sizes, int n_in,
                              void* d_out, int out_size, void* d_ws, size_t ws_size,
                              hipStream_t stream) {
  (void)in_sizes; (void)n_in; (void)out_size; (void)ws_size;
  const float* feats = (const float*)d_in[0];
  const int*   bidx  = (const int*)  d_in[1];
  const float* Wc    = (const float*)d_in[2];
  const float* bc    = (const float*)d_in[3];
  const float* W1    = (const float*)d_in[4];
  const float* b1    = (const float*)d_in[5];
  const float* gamma = (const float*)d_in[6];
  const float* beta  = (const float*)d_in[7];
  const float* W2    = (const float*)d_in[8];
  const float* b2    = (const float*)d_in[9];
  float* out = (float*)d_out;

  unsigned* wHi  = (unsigned*)d_ws;            // 65536 u32 (256 KB)
  unsigned* wLo  = wHi + 65536;                // 65536 u32
  unsigned* gKey = wLo + 65536;                // 32768 u32 (pooled keys)
  float*    h    = (float*)(gKey + 32768);     // 64*1024
  float*    h2   = h + BATCH * D1;             // 64*1024

  // pk 128K + B double-buffer 128K + A 8K = 264 KB dynamic LDS
  size_t smemA = (size_t)(32768 + 4 * 8192 + 2 * 1024) * sizeof(unsigned);

  k0_prep  <<<256, 256, 0,     stream>>>(Wc, wHi, wLo, gKey);
  gemm_pool<<<128, 256, smemA, stream>>>(feats, bidx, bc, wHi, wLo, gKey);
  fc1_wmma <<<32,  256, 0,     stream>>>(gKey, W1, b1, h);
  bn_relu  <<<4,   256, 0,     stream>>>(h, gamma, beta, h2);
  fc2_wmma <<<8,   256, 0,     stream>>>(h2, W2, b2, out);
}
